// GPSA_59820304499342
// MI455X (gfx1250) — matrix operand order repaired and checked by reading, not compiled, not run on hardware
//
#include <hip/hip_runtime.h>
#include <hip/hip_bf16.h>

typedef __bf16 bf16;
typedef bf16  v8bf  __attribute__((ext_vector_type(8)));
typedef bf16  v16bf __attribute__((ext_vector_type(16)));
typedef float v8f   __attribute__((ext_vector_type(8)));

#define B_    8
#define N_    784
#define D_    768
#define H_    16
#define HD_   48
#define NPAD  800
#define HDPAD 64

static __device__ __forceinline__ v8bf  ld8(const bf16* p)  { return *(const v8bf*)p; }
static __device__ __forceinline__ v16bf ld16(const bf16* p) { return *(const v16bf*)p; }
static __device__ __forceinline__ v16bf cat16(v8bf a, v8bf b) {
  return __builtin_shufflevector(a, b, 0,1,2,3,4,5,6,7,8,9,10,11,12,13,14,15);
}
static __device__ __forceinline__ v8f wmma_bf16(v16bf a, v16bf b, v8f c) {
  return __builtin_amdgcn_wmma_f32_16x16x32_bf16(false, a, false, b, (short)0, c, false, false);
}
static __device__ __forceinline__ float rmax16(float v) {
  v = fmaxf(v, __shfl_xor(v, 1)); v = fmaxf(v, __shfl_xor(v, 2));
  v = fmaxf(v, __shfl_xor(v, 4)); v = fmaxf(v, __shfl_xor(v, 8));
  return v;
}
static __device__ __forceinline__ float rsum16(float v) {
  v += __shfl_xor(v, 1); v += __shfl_xor(v, 2);
  v += __shfl_xor(v, 4); v += __shfl_xor(v, 8);
  return v;
}
static __device__ __forceinline__ float rmax32(float v) {
  v = fmaxf(v, __shfl_xor(v, 16)); return rmax16(v);
}
static __device__ __forceinline__ float rsum32(float v) {
  v += __shfl_xor(v, 16); return rsum16(v);
}

// ---------------------------------------------------------------- utilities
__global__ void cvt_f32_bf16(const float* __restrict__ in, bf16* __restrict__ out, int n) {
  for (int i = blockIdx.x * blockDim.x + threadIdx.x; i < n; i += gridDim.x * blockDim.x)
    out[i] = (bf16)in[i];
}
__global__ void zero_u32(unsigned int* __restrict__ p, long n) {
  for (long i = blockIdx.x * (long)blockDim.x + threadIdx.x; i < n; i += (long)gridDim.x * blockDim.x)
    p[i] = 0u;
}

// --------------------------------------------------- positional softmax rows
// pos[h][n][m] (m padded to 800 with zeros), bf16, softmax over m
__global__ void pos_softmax(const float* __restrict__ w_pos, const float* __restrict__ b_pos,
                            bf16* __restrict__ pos) {
  __shared__ float buf[4][NPAD];
  const int lane = threadIdx.x & 31, wave = threadIdx.x >> 5;
  const int row = blockIdx.x * 4 + wave;           // 0 .. H_*N_-1 (grid exact)
  const int h = row / N_, n = row % N_;
  const float w0 = w_pos[h * 3 + 0], w1 = w_pos[h * 3 + 1], w2 = w_pos[h * 3 + 2];
  const float bp = b_pos[h];
  const float cxn = (float)(n % 28), cyn = (float)(n / 28);
  float mx = -1e30f;
  for (int m = lane; m < N_; m += 32) {
    float cx = (float)(m % 28), cy = (float)(m / 28);
    float dx = cx - cxn, dy = cy - cyn;
    float lg = w0 * dx + w1 * dy + w2 * (dx * dx + dy * dy) + bp;
    buf[wave][m] = lg;
    mx = fmaxf(mx, lg);
  }
  mx = rmax32(mx);
  float sm = 0.f;
  for (int m = lane; m < N_; m += 32) {
    float e = __expf(buf[wave][m] - mx);
    buf[wave][m] = e;
    sm += e;
  }
  sm = rsum32(sm);
  const float inv = 1.f / sm;
  bf16* dst = pos + ((long)h * N_ + n) * NPAD;
  for (int m = lane; m < NPAD; m += 32)
    dst[m] = (bf16)((m < N_) ? buf[wave][m] * inv : 0.f);
}

// ----------------------------------------------------------- generic WMMA GEMM
// C[M,Nc] = A[M,K] * Bw[Nc,K]^T ; per wave: one 32x64 output tile
//   (2 A subtiles x 4 B subtiles -> 8 WMMAs per 32-k step)
// mode 0: scatter to q/k buffers  [B,H,NPAD,HDPAD] bf16
// mode 1: scatter to v-transposed [B,H,HD,NPAD]    bf16
// mode 2: f32 out + bias          [M,Nc]
__global__ void gemm_bf16_wmma(const bf16* __restrict__ A, const bf16* __restrict__ Bw,
                               int K, int Nc, int mode,
                               bf16* qbuf, bf16* kbuf, bf16* vtbuf,
                               float* outp, const float* bias) {
  const int lane = threadIdx.x & 31, wave = threadIdx.x >> 5;
  const int half = lane >> 4, ln = lane & 15;
  const int m0 = blockIdx.x * 32;
  const int n0 = (blockIdx.y * 4 + wave) * 64;
  if (n0 >= Nc) return;

  v8f acc[2][4] = {};
  const bf16* arow0 = A + (long)(m0 + ln) * K;
  const bf16* arow1 = A + (long)(m0 + 16 + ln) * K;
  for (int k0 = 0; k0 < K; k0 += 32) {
    if (k0 + 32 < K) {                       // keep L2 -> WGP path ahead
      __builtin_prefetch(arow0 + k0 + 32, 0, 1);
      __builtin_prefetch(arow1 + k0 + 32, 0, 1);
    }
    v16bf a0 = cat16(ld8(arow0 + k0 + 8 * half), ld8(arow0 + k0 + 16 + 8 * half));
    v16bf a1 = cat16(ld8(arow1 + k0 + 8 * half), ld8(arow1 + k0 + 16 + 8 * half));
#pragma unroll
    for (int t = 0; t < 4; ++t) {
      const bf16* brow = Bw + (long)(n0 + t * 16 + ln) * K + k0;
      v16bf b = cat16(ld8(brow + 8 * half), ld8(brow + 16 + 8 * half));
      acc[0][t] = wmma_bf16(a0, b, acc[0][t]);
      acc[1][t] = wmma_bf16(a1, b, acc[1][t]);
    }
  }
#pragma unroll
  for (int mt = 0; mt < 2; ++mt) {
#pragma unroll
    for (int t = 0; t < 4; ++t) {
#pragma unroll
      for (int r = 0; r < 8; ++r) {
        const int i = m0 + mt * 16 + r + 8 * half;
        const int j = n0 + t * 16 + ln;
        const float val = acc[mt][t][r];
        if (mode == 0) {
          const int b = i / N_, n = i % N_;
          const int sel = (j >= D_) ? 1 : 0;
          const int jj = j - sel * D_;
          const int h = jj / HD_, e = jj % HD_;
          bf16* dst = sel ? kbuf : qbuf;
          dst[(((long)b * H_ + h) * NPAD + n) * HDPAD + e] = (bf16)val;
        } else if (mode == 1) {
          const int b = i / N_, n = i % N_;
          const int h = j / HD_, e = j % HD_;
          vtbuf[(((long)b * H_ + h) * HD_ + e) * NPAD + n] = (bf16)val;
        } else {
          outp[(long)i * D_ + j] = val + bias[j];
        }
      }
    }
  }
}

// -------------------------------------------------- fused gated attention
// per wave: one 16-row query tile of one (b,h); dual accumulators:
//   O_c = flash online-softmax(QK^T*scale) @ V   (normalized by l at end)
//   O_p = pos_score @ V                          (already normalized)
// y = (1-g)*O_c/l + g*O_p
__global__ void gpsa_attn(const bf16* __restrict__ q, const bf16* __restrict__ k,
                          const bf16* __restrict__ vt, const bf16* __restrict__ pos,
                          const float* __restrict__ gating, bf16* __restrict__ yatt) {
  __shared__ bf16 lds_p[4][16 * 32];
  const int lane = threadIdx.x & 31, wave = threadIdx.x >> 5;
  const int half = lane >> 4, ln = lane & 15;
  const int bh = blockIdx.x;          // b*H + h
  const int b = bh >> 4, h = bh & 15;
  const int qt = blockIdx.y * 4 + wave;
  if (qt >= 49) return;
  const int n0 = qt * 16;

  const float g = 1.f / (1.f + __expf(-gating[h]));
  const float scale = 0.14433756729740643f;   // 48^-0.5

  const bf16* qrow = q + ((long)bh * NPAD + n0 + ln) * HDPAD;
  const v16bf aq0 = cat16(ld8(qrow + 8 * half),      ld8(qrow + 16 + 8 * half));
  const v16bf aq1 = cat16(ld8(qrow + 32 + 8 * half), ld8(qrow + 48 + 8 * half));

  v8f oc[3] = {}, op[3] = {};
  float mrow[8], lrow[8];
#pragma unroll
  for (int r = 0; r < 8; ++r) { mrow[r] = -1e30f; lrow[r] = 0.f; }

  for (int j0 = 0; j0 < NPAD; j0 += 32) {
    // ---- content scores S[16,32]
    v8f s[2] = {};
#pragma unroll
    for (int t = 0; t < 2; ++t) {
      const bf16* krow = k + ((long)bh * NPAD + j0 + t * 16 + ln) * HDPAD;
      if (j0 + 32 < NPAD) __builtin_prefetch(krow + 32 * HDPAD, 0, 1);
      s[t] = wmma_bf16(aq0, cat16(ld8(krow + 8 * half), ld8(krow + 16 + 8 * half)), s[t]);
      s[t] = wmma_bf16(aq1, cat16(ld8(krow + 32 + 8 * half), ld8(krow + 48 + 8 * half)), s[t]);
    }
#pragma unroll
    for (int t = 0; t < 2; ++t) {
      const int col = j0 + t * 16 + ln;
#pragma unroll
      for (int r = 0; r < 8; ++r) {
        float v = s[t][r] * scale;
        if (col >= N_) v = -1e30f;
        s[t][r] = v;
      }
    }
    // ---- online softmax stats (per row, reduced over 16 lanes of the half)
    float alpha[8];
#pragma unroll
    for (int r = 0; r < 8; ++r) {
      float mc = rmax16(fmaxf(s[0][r], s[1][r]));
      float mn = fmaxf(mrow[r], mc);
      alpha[r] = __expf(mrow[r] - mn);
      mrow[r] = mn;
    }
#pragma unroll
    for (int t = 0; t < 2; ++t)
#pragma unroll
      for (int r = 0; r < 8; ++r)
        s[t][r] = __expf(s[t][r] - mrow[r]);
#pragma unroll
    for (int r = 0; r < 8; ++r) {
      float rs = rsum16(s[0][r] + s[1][r]);
      lrow[r] = lrow[r] * alpha[r] + rs;
    }
#pragma unroll
    for (int t = 0; t < 3; ++t)
#pragma unroll
      for (int r = 0; r < 8; ++r)
        oc[t][r] *= alpha[r];

    // ---- C-layout -> A-layout for P through LDS (bf16)
#pragma unroll
    for (int t = 0; t < 2; ++t)
#pragma unroll
      for (int r = 0; r < 8; ++r)
        lds_p[wave][(r + 8 * half) * 32 + t * 16 + ln] = (bf16)s[t][r];
    asm volatile("s_wait_dscnt 0x0" ::: "memory");
    const bf16* prow = &lds_p[wave][ln * 32];
    const v16bf ap = cat16(*(const v8bf*)(prow + 8 * half),
                           *(const v8bf*)(prow + 16 + 8 * half));

    // ---- positional scores, pre-normalized bf16, A-layout direct from global
    const bf16* posrow = pos + ((long)h * N_ + n0 + ln) * NPAD + j0;
    const v16bf apos = cat16(ld8(posrow + 8 * half), ld8(posrow + 16 + 8 * half));

    // ---- accumulate both P@V and pos@V
#pragma unroll
    for (int t = 0; t < 3; ++t) {
      const bf16* vrow = vt + ((long)bh * HD_ + t * 16 + ln) * NPAD + j0;
      const v16bf bv = cat16(ld8(vrow + 8 * half), ld8(vrow + 16 + 8 * half));
      oc[t] = wmma_bf16(ap, bv, oc[t]);
      op[t] = wmma_bf16(apos, bv, op[t]);
    }
  }

  // ---- epilogue: gated blend, write y [B,N,D] bf16
#pragma unroll
  for (int t = 0; t < 3; ++t)
#pragma unroll
    for (int r = 0; r < 8; ++r) {
      const int m = r + 8 * half;
      const float val = (1.f - g) * oc[t][r] / lrow[r] + g * op[t][r];
      yatt[((long)b * N_ + n0 + m) * D_ + h * HD_ + t * 16 + ln] = (bf16)val;
    }
}

// ------------------------------------------------------------------ launcher
extern "C" void kernel_launch(void* const* d_in, const int* in_sizes, int n_in,
                              void* d_out, int out_size, void* d_ws, size_t ws_size,
                              hipStream_t stream) {
  (void)in_sizes; (void)n_in; (void)out_size; (void)ws_size;
  const float* x      = (const float*)d_in[0];
  const float* w_qk   = (const float*)d_in[1];
  const float* w_v    = (const float*)d_in[2];
  const float* w_proj = (const float*)d_in[3];
  const float* b_proj = (const float*)d_in[4];
  const float* w_pos  = (const float*)d_in[5];
  const float* b_pos  = (const float*)d_in[6];
  const float* gating = (const float*)d_in[7];
  float* outp = (float*)d_out;

  char* ws = (char*)d_ws;
  bf16* xb    = (bf16*)(ws + 0);          //  9,633,792  x bf16 [B*N, D]
  bf16* wqkb  = (bf16*)(ws + 9633792);    //  2,359,296  w_qk bf16 [2D, D]
  bf16* wvb   = (bf16*)(ws + 11993088);   //  1,179,648  w_v bf16
  bf16* wpb   = (bf16*)(ws + 13172736);   //  1,179,648  w_proj bf16
  bf16* qb    = (bf16*)(ws + 14352384);   // 13,107,200  q [B,H,800,64]
  bf16* kb    = (bf16*)(ws + 27459584);   // 13,107,200  k [B,H,800,64]
  bf16* vtb   = (bf16*)(ws + 40566784);   //  9,830,400  vT [B,H,48,800]
  bf16* posb  = (bf16*)(ws + 50397184);   // 20,070,400  pos [H,784,800]
  bf16* yb    = (bf16*)(ws + 70467584);   //  9,633,792  attn out bf16 [B*N, D]

  // 1) convert inputs to bf16
  cvt_f32_bf16<<<4096, 256, 0, stream>>>(x,      xb,   B_ * N_ * D_);
  cvt_f32_bf16<<<2048, 256, 0, stream>>>(w_qk,   wqkb, 2 * D_ * D_);
  cvt_f32_bf16<<<1024, 256, 0, stream>>>(w_v,    wvb,  D_ * D_);
  cvt_f32_bf16<<<1024, 256, 0, stream>>>(w_proj, wpb,  D_ * D_);

  // 2) zero q/k/vT pads (contiguous region, 36,044,800 B)
  zero_u32<<<8192, 256, 0, stream>>>((unsigned int*)qb, 36044800L / 4);

  // 3) positional softmax (batch independent)
  pos_softmax<<<(H_ * N_) / 4, 128, 0, stream>>>(w_pos, b_pos, posb);

  // 4) q/k projection  (Nc = 1536, 32x64 tiles)
  gemm_bf16_wmma<<<dim3(196, 6), 128, 0, stream>>>(xb, wqkb, D_, 2 * D_, 0,
                                                   qb, kb, nullptr, nullptr, nullptr);
  // 5) v projection, transposed store (Nc = 768)
  gemm_bf16_wmma<<<dim3(196, 3), 128, 0, stream>>>(xb, wvb, D_, D_, 1,
                                                   nullptr, nullptr, vtb, nullptr, nullptr);
  // 6) fused gated attention
  gpsa_attn<<<dim3(B_ * H_, 13), 128, 0, stream>>>(qb, kb, vtb, posb, gating, yb);

  // 7) output projection + bias (f32 out)
  gemm_bf16_wmma<<<dim3(196, 3), 128, 0, stream>>>(yb, wpb, D_, D_, 2,
                                                   nullptr, nullptr, nullptr, outp, b_proj);
}
